// Attention_65446711656944
// MI455X (gfx1250) — matrix operand order repaired and checked by reading, not compiled, not run on hardware
//
#include <hip/hip_runtime.h>

// Flash attention: N=M=8192, D=DV=256, fp32 in/out, bf16 WMMA math (f32 accum).
// Per wave: 16 query rows; online softmax over 256 chunks of 32 keys.
// S^T = K x Q^T (wmma), O^T = V^T x P^T (wmma).
// Staging latency hidden by global_prefetch of chunk k+1 issued before compute.

#define NQ    8192
#define NK    8192
#define DHEAD 256
#define DVAL  256

#define WAVES 4
#define BLOCK (WAVES * 32)   // 128 threads
#define QPW   16             // query rows per wave
#define QPB   (WAVES * QPW)  // 64 query rows per block
#define CK    32             // keys per chunk
#define NCH   (NK / CK)      // 256 chunks
#define KP    264            // K LDS row pitch (bf16 elems): 256 + 8 pad
#define VP    40             // V LDS row pitch (bf16 elems): 32 keys + 8 pad

typedef __attribute__((ext_vector_type(16))) __bf16          v16b;
typedef __attribute__((ext_vector_type(16))) unsigned short  v16us;
typedef __attribute__((ext_vector_type(8)))  float           v8f;

union BFrag { v16us u; v16b b; uint4 q[2]; unsigned d[8]; };
union FU { float f; unsigned u; };

// pack two f32 -> {hi.bf16, lo.bf16} in one v_perm_b32 (truncating convert)
static __device__ __forceinline__ unsigned pack_bf16(float lo, float hi) {
    FU a, b; a.f = lo; b.f = hi;
    return __builtin_amdgcn_perm(b.u, a.u, 0x07060302u);
}

static __device__ __forceinline__ v8f wmma_bf16(const BFrag& a, const BFrag& b, v8f c) {
    return __builtin_amdgcn_wmma_f32_16x16x32_bf16(false, a.b, false, b.b,
                                                   (short)0, c, false, false);
}

__global__ __launch_bounds__(BLOCK, 2)
void fa_fwd_kernel(const float* __restrict__ Q, const float* __restrict__ K,
                   const float* __restrict__ V, float* __restrict__ O)
{
    __shared__ __align__(16) unsigned short kl[CK * KP];    // K chunk bf16 [key][d]
    __shared__ __align__(16) unsigned short vl[DVAL * VP];  // V chunk bf16 [dv][key]

    const int tid  = threadIdx.x;
    const int wave = tid >> 5;
    const int lane = tid & 31;
    const int h    = lane >> 4;   // lane half
    const int ln   = lane & 15;
    const int qrow = blockIdx.x * QPB + wave * QPW + ln;

    // ---- Q fragments, pre-scaled by 1/sqrt(d)*log2(e) so scores land in exp2 domain
    // B-layout: lane = query column ln, elems = d 32*ks + 16*h + e
    const float qs = 0.0625f * 1.44269504f;
    BFrag qf[8];
    {
        const float* qp = Q + (size_t)qrow * DHEAD + 16 * h;
        #pragma unroll
        for (int ks = 0; ks < 8; ++ks) {
            const float4* p = (const float4*)(qp + ks * 32);
            float4 f0 = p[0], f1 = p[1], f2 = p[2], f3 = p[3];
            qf[ks].d[0] = pack_bf16(f0.x * qs, f0.y * qs);
            qf[ks].d[1] = pack_bf16(f0.z * qs, f0.w * qs);
            qf[ks].d[2] = pack_bf16(f1.x * qs, f1.y * qs);
            qf[ks].d[3] = pack_bf16(f1.z * qs, f1.w * qs);
            qf[ks].d[4] = pack_bf16(f2.x * qs, f2.y * qs);
            qf[ks].d[5] = pack_bf16(f2.z * qs, f2.w * qs);
            qf[ks].d[6] = pack_bf16(f3.x * qs, f3.y * qs);
            qf[ks].d[7] = pack_bf16(f3.z * qs, f3.w * qs);
        }
    }

    v8f acc[16];                       // O^T tiles: dv = 16*t + r + 8*h, query = ln
    #pragma unroll
    for (int t = 0; t < 16; ++t) acc[t] = v8f{};
    float m_run = -INFINITY, l_run = 0.0f;

    // staging: idx = i*128+tid -> d4 = tid&63 (invariant), key = 2*i + (tid>>6)
    const int d4   = tid & 63;
    const int krow = tid >> 6;
    const float* kgp = K + (size_t)krow * DHEAD + d4 * 4;
    const float* vgp = V + (size_t)krow * DVAL + d4 * 4;
    unsigned short* klp  = &kl[krow * KP + d4 * 4];
    unsigned short* vlp  = &vl[(d4 * 4) * VP + krow];
    const bool pf_lane = ((tid & 7) == 0);   // one requester per 128B line

    for (int kc = 0; kc < NCH; ++kc) {
        __syncthreads();               // all waves done reading LDS (prev chunk)

        // ---- stage chunk kc: fp32 global -> bf16 LDS (K row-major, V transposed)
        #pragma unroll
        for (int i = 0; i < 16; ++i) {
            float4 fk = *(const float4*)(kgp + i * 2 * DHEAD);
            float4 fv = *(const float4*)(vgp + i * 2 * DVAL);
            *(uint2*)(klp + i * 2 * KP) =
                make_uint2(pack_bf16(fk.x, fk.y), pack_bf16(fk.z, fk.w));
            unsigned v01 = pack_bf16(fv.x, fv.y);
            unsigned v23 = pack_bf16(fv.z, fv.w);
            vlp[0 * VP + 2 * i] = (unsigned short)(v01);
            vlp[1 * VP + 2 * i] = (unsigned short)(v01 >> 16);
            vlp[2 * VP + 2 * i] = (unsigned short)(v23);
            vlp[3 * VP + 2 * i] = (unsigned short)(v23 >> 16);
        }
        __syncthreads();
        kgp += CK * DHEAD;
        vgp += CK * DVAL;

        // ---- prefetch chunk kc+1 into cache while we compute (no regs, no waits)
        if (kc + 1 < NCH && pf_lane) {
            #pragma unroll
            for (int i = 0; i < 16; ++i) {
                __builtin_prefetch(kgp + i * 2 * DHEAD, 0, 1);
                __builtin_prefetch(vgp + i * 2 * DVAL, 0, 1);
            }
        }

        // ---- S^T = K_tile x Q^T : two 16(key) x 16(query) tiles (exp2 domain)
        v8f s0 = v8f{}, s1 = v8f{};
        #pragma unroll
        for (int ks = 0; ks < 8; ++ks) {
            const unsigned short* kr0 = &kl[ln * KP + 32 * ks + 16 * h];
            const unsigned short* kr1 = &kl[(16 + ln) * KP + 32 * ks + 16 * h];
            BFrag a0, a1;
            a0.q[0] = *(const uint4*)kr0;  a0.q[1] = *(const uint4*)(kr0 + 8);
            a1.q[0] = *(const uint4*)kr1;  a1.q[1] = *(const uint4*)(kr1 + 8);
            s0 = wmma_bf16(a0, qf[ks], s0);
            s1 = wmma_bf16(a1, qf[ks], s1);
        }

        // ---- online softmax (lane: query ln; 8 keys per tile in this lane half)
        float mc = -INFINITY;
        #pragma unroll
        for (int r = 0; r < 8; ++r) mc = fmaxf(mc, fmaxf(s0[r], s1[r]));
        mc = fmaxf(mc, __shfl_xor(mc, 16, 32));

        if (__ballot(mc > m_run)) {    // wave-uniform: rescale only when max grows
            float m_new = fmaxf(m_run, mc);
            float alpha = exp2f(m_run - m_new);
            #pragma unroll
            for (int t = 0; t < 16; ++t) acc[t] *= alpha;
            l_run *= alpha;
            m_run = m_new;
        }

        float p0[8], p1[8], lc = 0.0f;
        #pragma unroll
        for (int r = 0; r < 8; ++r) {
            p0[r] = exp2f(s0[r] - m_run);
            p1[r] = exp2f(s1[r] - m_run);
            lc += p0[r] + p1[r];
        }
        lc += __shfl_xor(lc, 16, 32);
        l_run += lc;

        // ---- P^T C-layout -> B-layout bf16 frag (32 keys x 16 queries)
        BFrag pf;
        #pragma unroll
        for (int r = 0; r < 8; r += 2) {
            float a0 = __shfl_xor(p0[r], 16, 32), a1 = __shfl_xor(p0[r + 1], 16, 32);
            float b0 = __shfl_xor(p1[r], 16, 32), b1 = __shfl_xor(p1[r + 1], 16, 32);
            float lo0 = (h == 0) ? p0[r]     : b0;   // key 16h + r
            float lo1 = (h == 0) ? p0[r + 1] : b1;   // key 16h + r+1
            float hi0 = (h == 0) ? a0 : p1[r];       // key 16h + 8 + r
            float hi1 = (h == 0) ? a1 : p1[r + 1];   // key 16h + 8 + r+1
            pf.d[r / 2]     = pack_bf16(lo0, lo1);
            pf.d[4 + r / 2] = pack_bf16(hi0, hi1);
        }

        // ---- O^T += V^T x P^T : 16 dv tiles
        #pragma unroll
        for (int t = 0; t < 16; ++t) {
            const unsigned short* vr = &vl[(16 * t + ln) * VP + 16 * h];
            BFrag va;
            va.q[0] = *(const uint4*)vr;        // keys 16h..16h+7
            va.q[1] = *(const uint4*)(vr + 8);  // keys 16h+8..16h+15
            acc[t] = wmma_bf16(va, pf, acc[t]);
        }
    }

    // ---- normalize + store: lane holds query qrow, dv = 16t + 8h + r (contiguous)
    float inv = 1.0f / l_run;
    #pragma unroll
    for (int t = 0; t < 16; ++t) {
        v8f o = acc[t] * inv;
        float* op = O + (size_t)qrow * DVAL + 16 * t + 8 * h;
        *(float4*)(op)     = make_float4(o[0], o[1], o[2], o[3]);
        *(float4*)(op + 4) = make_float4(o[4], o[5], o[6], o[7]);
    }
}

extern "C" void kernel_launch(void* const* d_in, const int* in_sizes, int n_in,
                              void* d_out, int out_size, void* d_ws, size_t ws_size,
                              hipStream_t stream) {
    (void)in_sizes; (void)n_in; (void)out_size; (void)d_ws; (void)ws_size;
    const float* Q = (const float*)d_in[0];
    const float* K = (const float*)d_in[1];
    const float* V = (const float*)d_in[2];
    float*       O = (float*)d_out;
    dim3 grid(NQ / QPB);   // 128 blocks
    dim3 block(BLOCK);     // 128 threads = 4 waves
    fa_fwd_kernel<<<grid, block, 0, stream>>>(Q, K, V, O);
}
